// GAT_30382598652184
// MI455X (gfx1250) — matrix operand order repaired and checked by reading, not compiled, not run on hardware
//
#include <hip/hip_runtime.h>
#include <hip/hip_bf16.h>

#define NN 8192      // total nodes
#define NU 4096      // users
#define DD 64        // feature dim
#define ALPHA 0.2f

typedef __attribute__((ext_vector_type(16))) _Float16 v16h;
typedef __attribute__((ext_vector_type(8)))  _Float16 v8h;
typedef __attribute__((ext_vector_type(8)))  float    v8f;

// ---------------------------------------------------------------------------
// Kernel 1: per-node projection  Wh = x @ W^T + b,  src = Wh@a[:D], dst = Wh@a[D:]
// Stores WhT as f16 in [d][j] layout so attention B-tiles are contiguous loads.
// ---------------------------------------------------------------------------
__global__ __launch_bounds__(256) void gat_project(
    const float* __restrict__ xa,   // rows j <  NU read from xa[j]
    const float* __restrict__ xb,   // rows j >= NU read from xb[j-NU]
    const float* __restrict__ W,    // D x D row-major (d,k)
    const float* __restrict__ bias, // D
    const float* __restrict__ avec, // 2D: [0..D)=a_src, [D..2D)=a_dst
    _Float16* __restrict__ WhT,     // D x NN
    float* __restrict__ srcv,       // NN
    float* __restrict__ dstv)       // NN
{
    __shared__ float Ws[DD * DD];
    __shared__ float bs[DD], as0[DD], as1[DD];
    __shared__ float xs[4][DD];
    __shared__ float Whs[4][DD];

    const int tid  = threadIdx.x;
    const int jloc = tid >> 6;       // 0..3
    const int d    = tid & 63;
    const int j    = blockIdx.x * 4 + jloc;

    for (int i = tid; i < DD * DD; i += 256) Ws[i] = W[i];
    if (tid < DD) { bs[tid] = bias[tid]; as0[tid] = avec[tid]; as1[tid] = avec[DD + tid]; }

    const float* xrow = (j < NU) ? (xa + (size_t)j * DD) : (xb + (size_t)(j - NU) * DD);
    xs[jloc][d] = xrow[d];
    __syncthreads();

    float acc = bs[d];
    #pragma unroll
    for (int k = 0; k < DD; ++k) acc += xs[jloc][k] * Ws[d * DD + k];

    WhT[(size_t)d * NN + j] = (_Float16)acc;
    Whs[jloc][d] = acc;
    __syncthreads();

    if (d == 0) {
        float s = 0.f, t = 0.f;
        #pragma unroll
        for (int k = 0; k < DD; ++k) { s += Whs[jloc][k] * as0[k]; t += Whs[jloc][k] * as1[k]; }
        srcv[j] = s;
        dstv[j] = t;
    }
}

// ---------------------------------------------------------------------------
// Kernel 2: fused masked attention + softmax + att@Wh (+relu).
// One wave owns a 16-row panel, streams all 8192 columns once.
// MODE 0: read raw int32 adj, ALSO emit packed 1-bit adjacency (8 MB).
// MODE 1: read packed adjacency bits (32x less adj traffic, L2-resident).
// MODE 2: read raw adj only (fallback when workspace too small for bits).
//
// Packed word layout (lane-local, writer==reader by construction):
//   bits [15:0]  = lane<16 slots e=0..15  (columns jb+{0..7,16..23})
//   bits [31:16] = lane>=16 slots e=0..15 (columns jb+{8..15,24..31})
// -> writer: one global_store_b16 per lane, no cross-lane shuffle in the loop.
//
// Masking feeds the exp INPUT (lr = adj ? lr : -inf) so v_exp_f32 stays
// unconditional -> no divergent per-element branches.
// ---------------------------------------------------------------------------
template <int MODE>
__global__ __launch_bounds__(128) void gat_attention(
    const int* __restrict__ adj,              // NN x NN (MODE 0/2)
    const unsigned* __restrict__ bits_in,     // NN x NN/32 (MODE 1)
    unsigned short* __restrict__ bits_out,    // NN x NN/16 halfwords (MODE 0)
    const _Float16* __restrict__ WhT,         // D x NN
    const float* __restrict__ srcv,           // NN
    const float* __restrict__ dstv,           // NN
    float* __restrict__ x_out)                // NN x D = relu(softmax(e) @ Wh)
{
    const int wave  = threadIdx.x >> 5;
    const int lane  = threadIdx.x & 31;
    const int panel = blockIdx.x * 4 + wave;   // 0..511
    const int row0  = panel * 16;
    const int m     = lane & 15;               // A-matrix row for this lane
    const int row   = row0 + m;
    const bool lo   = (lane < 16);
    const int koff0 = lo ? 0 : 8;              // first 8-wide K chunk this lane owns
    // A layout (16-bit 16x32): lanes 0-15 hold K={0..7,16..23}, lanes 16-31 K={8..15,24..31}
    // element e -> K slot: e<8 ? koff0+e : koff0+16+(e-8)

    const float s_i = srcv[row];
    const float NEGINF = -__builtin_inff();
    const int* adjrow = adj + (size_t)row * NN;
    const size_t wrow = (size_t)row * (NN / 32);           // 32-bit word index base
    // per-lane halfword slot: word*2 + (lo?0:1)
    unsigned short* hw_out = bits_out + wrow * 2 + (lo ? 0 : 1);
    // B pointer: lane holds N = n0 + (lane&15), K chunks {+0,+16} (lanes<16) or {+8,+24}
    const _Float16* wbase = WhT + (size_t)m * NN + (lo ? 0 : 8);

    v8f acc[4] = {};                    // 16x64 f32 output panel (4 N-slices of 16)
    float ps0 = 0.f, ps1 = 0.f, ps2 = 0.f, ps3 = 0.f;   // partial row sums of p

    for (int jb = 0; jb < NN; jb += 32) {
        int qa[16];
        unsigned w = 0;
        if constexpr (MODE != 1) {
            __builtin_prefetch(adjrow + jb + 2048, 0, 1);   // stream adj ~8KB ahead
            const int4* aj0 = (const int4*)(adjrow + jb + koff0);
            const int4* aj1 = (const int4*)(adjrow + jb + koff0 + 16);
            int4 q0 = aj0[0], q1 = aj0[1], q2 = aj1[0], q3 = aj1[1];
            qa[0]=q0.x; qa[1]=q0.y; qa[2]=q0.z;  qa[3]=q0.w;
            qa[4]=q1.x; qa[5]=q1.y; qa[6]=q1.z;  qa[7]=q1.w;
            qa[8]=q2.x; qa[9]=q2.y; qa[10]=q2.z; qa[11]=q2.w;
            qa[12]=q3.x;qa[13]=q3.y;qa[14]=q3.z; qa[15]=q3.w;
        } else {
            w = bits_in[wrow + (jb >> 5)] >> (lo ? 0 : 16); // this lane's 16 slot bits
        }

        const float4* dp0 = (const float4*)(dstv + jb + koff0);
        const float4* dp1 = (const float4*)(dstv + jb + koff0 + 16);
        float4 d0 = dp0[0], d1 = dp0[1], d2 = dp1[0], d3 = dp1[1];
        const float da[16] = { d0.x,d0.y,d0.z,d0.w, d1.x,d1.y,d1.z,d1.w,
                               d2.x,d2.y,d2.z,d2.w, d3.x,d3.y,d3.z,d3.w };

        // ---- p = exp( adj ? leakyrelu(src_i + dst_j) : -inf )  -- branchless ----
        v16h A;
        float pe[16];
        #pragma unroll
        for (int e = 0; e < 16; ++e) {
            float ev = s_i + da[e];
            float lr = fmaxf(ev, ALPHA * ev);          // leaky relu (alpha < 1)
            bool on;
            if constexpr (MODE == 1) on = (w & (1u << e)) != 0;
            else                     on = qa[e] != 0;
            lr = on ? lr : NEGINF;                     // single v_cndmask, pre-exp
            float p  = __expf(lr);                     // exp(-inf) == 0
            pe[e] = p;
            A[e] = (_Float16)p;
        }
        #pragma unroll
        for (int e = 0; e < 16; e += 4) {              // 4 independent sum chains
            ps0 += pe[e];  ps1 += pe[e + 1];  ps2 += pe[e + 2];  ps3 += pe[e + 3];
        }

        if constexpr (MODE == 0) {
            // adj values are exactly {0,1}: OR the bits in a log2-depth tree
            // (15 v_lshl_or ops, depth 4 -- no serial 16-deep chain)
            unsigned t[16];
            #pragma unroll
            for (int e = 0; e < 16; ++e) t[e] = (unsigned)qa[e];
            #pragma unroll
            for (int s = 1; s < 16; s <<= 1) {
                #pragma unroll
                for (int e = 0; e < 16; e += 2 * s) t[e] |= t[e + s] << s;
            }
            // lane-local halfword store: no shuffle, no LDS wait, 8 MB total
            hw_out[(size_t)(jb >> 5) * 2] = (unsigned short)t[0];
        }

        // ---- 4 WMMAs: acc[t] += A(16x32) @ WhT-slice(32x16) ----
        #pragma unroll
        for (int t = 0; t < 4; ++t) {
            const _Float16* bp = wbase + (size_t)(16 * t) * NN + jb;
            v8h b_lo = *(const v8h*)(bp);
            v8h b_hi = *(const v8h*)(bp + 16);
            v16h B;
            #pragma unroll
            for (int e = 0; e < 8; ++e) { B[e] = b_lo[e]; B[e + 8] = b_hi[e]; }
            acc[t] = __builtin_amdgcn_wmma_f32_16x16x32_f16(
                false, A, false, B, (short)0, acc[t], false, false);
        }
    }

    // ---- row sums: lane l and l+16 cover complementary K halves of row (l&15) ----
    float psum  = (ps0 + ps1) + (ps2 + ps3);
    float total = psum + __shfl_xor(psum, 16, 32);
    const int rbase = lo ? 0 : 8;   // C layout: lanes 0-15 vgpr r -> M=r; lanes 16-31 -> M=8+r
    float inv[8];
    #pragma unroll
    for (int r = 0; r < 8; ++r) {
        float sr = __shfl(total, rbase + r, 32);
        inv[r] = __builtin_amdgcn_rcpf(sr);            // v_rcp_f32: softmax scale
    }

    // ---- normalize, relu, store ----
    const int ncol0 = lane & 15;
    #pragma unroll
    for (int t = 0; t < 4; ++t) {
        #pragma unroll
        for (int r = 0; r < 8; ++r) {
            float v = acc[t][r] * inv[r];
            v = v > 0.f ? v : 0.f;
            x_out[(size_t)(row0 + rbase + r) * DD + 16 * t + ncol0] = v;
        }
    }
}

// ---------------------------------------------------------------------------
// Kernel 3: final linear  out = x @ out_w^T + out_b  (rows already in u|it order)
// ---------------------------------------------------------------------------
__global__ __launch_bounds__(256) void final_linear(
    const float* __restrict__ x,    // NN x D
    const float* __restrict__ W,    // D x D
    const float* __restrict__ bias, // D
    float* __restrict__ out)        // NN x D
{
    __shared__ float Ws[DD * DD];
    __shared__ float bs[DD];
    __shared__ float xs[4][DD];

    const int tid  = threadIdx.x;
    const int jloc = tid >> 6;
    const int d    = tid & 63;
    const int j    = blockIdx.x * 4 + jloc;

    for (int i = tid; i < DD * DD; i += 256) Ws[i] = W[i];
    if (tid < DD) bs[tid] = bias[tid];
    xs[jloc][d] = x[(size_t)j * DD + d];
    __syncthreads();

    float acc = bs[d];
    #pragma unroll
    for (int k = 0; k < DD; ++k) acc += xs[jloc][k] * Ws[d * DD + k];
    out[(size_t)j * DD + d] = acc;
}

// ---------------------------------------------------------------------------
extern "C" void kernel_launch(void* const* d_in, const int* in_sizes, int n_in,
                              void* d_out, int out_size, void* d_ws, size_t ws_size,
                              hipStream_t stream) {
    const int*   adj      = (const int*)  d_in[0];
    const float* user_emb = (const float*)d_in[1];
    const float* item_emb = (const float*)d_in[2];
    const float* W0_w     = (const float*)d_in[3];
    const float* W0_b     = (const float*)d_in[4];
    const float* a0       = (const float*)d_in[5];
    const float* W1_w     = (const float*)d_in[6];
    const float* W1_b     = (const float*)d_in[7];
    const float* a1       = (const float*)d_in[8];
    const float* out_w    = (const float*)d_in[9];
    const float* out_b    = (const float*)d_in[10];
    float* out = (float*)d_out;

    // workspace layout
    char* ws = (char*)d_ws;
    float*     x1   = (float*)(ws);                                  // 2 MB
    float*     x2   = (float*)(ws + (size_t)NN * DD * 4);            // 2 MB
    _Float16*  WhT  = (_Float16*)(ws + (size_t)2 * NN * DD * 4);     // 1 MB
    float*     srcv = (float*)(ws + (size_t)2 * NN * DD * 4 + (size_t)DD * NN * 2);
    float*     dstv = srcv + NN;
    unsigned*  bits = (unsigned*)(dstv + NN);                        // 8 MB
    const size_t need_packed =
        (size_t)2 * NN * DD * 4 + (size_t)DD * NN * 2 + (size_t)2 * NN * 4
        + (size_t)NN * (NN / 32) * 4;
    const bool use_packed = ws_size >= need_packed;

    dim3 bProj(256), gProj(NN / 4);
    dim3 bAtt(128),  gAtt(NN / 16 / 4);   // 512 panels, 4 waves per block

    // layer 0 (streams raw adj once; optionally emits 1-bit packed adjacency)
    gat_project<<<gProj, bProj, 0, stream>>>(user_emb, item_emb, W0_w, W0_b, a0,
                                             WhT, srcv, dstv);
    if (use_packed)
        gat_attention<0><<<gAtt, bAtt, 0, stream>>>(adj, nullptr, (unsigned short*)bits,
                                                    WhT, srcv, dstv, x1);
    else
        gat_attention<2><<<gAtt, bAtt, 0, stream>>>(adj, nullptr, nullptr,
                                                    WhT, srcv, dstv, x1);
    // layer 1 (packed path reads 8 MB of bits instead of 256 MB of int32)
    gat_project<<<gProj, bProj, 0, stream>>>(x1, x1 + (size_t)NU * DD, W1_w, W1_b, a1,
                                             WhT, srcv, dstv);
    if (use_packed)
        gat_attention<1><<<gAtt, bAtt, 0, stream>>>(nullptr, bits, nullptr,
                                                    WhT, srcv, dstv, x2);
    else
        gat_attention<2><<<gAtt, bAtt, 0, stream>>>(adj, nullptr, nullptr,
                                                    WhT, srcv, dstv, x2);
    // output head
    final_linear<<<gProj, bProj, 0, stream>>>(x2, out_w, out_b, out);
}
